// Agent_74878459838639
// MI455X (gfx1250) — matrix operand order repaired and checked by reading, not compiled, not run on hardware
//
#include <hip/hip_runtime.h>
#include <hip/hip_bf16.h>
#include <math.h>

// ---------------------------------------------------------------------------
// Problem constants (from reference)
// ---------------------------------------------------------------------------
#define Bb    8192
#define Tt    48
#define DEMOn 2
#define LABn  73
#define Hn    128
#define NAn   10
#define NUn   64
#define FDn   128
#define NTHREADS 256

typedef __attribute__((ext_vector_type(16))) _Float16 v16h;
typedef __attribute__((ext_vector_type(8)))  _Float16 v8h;
typedef __attribute__((ext_vector_type(8)))  float    v8f;

// ---------------------------------------------------------------------------
// Dynamic LDS layout (bytes). Total 287744 B (~281 KB) < 320 KB/WGP.
// ---------------------------------------------------------------------------
#define OFF_WBHH   0        /* 24nt x 4kt frags, f16: 98304 B  */
#define OFF_WBIH   98304    /* 24nt x 3kt frags, f16: 73728 B  (reused for Wf) */
#define OFF_W1A    172032   /* 4nt x 5kt frags:       20480 B  */
#define OFF_W1B    192512   /* 4nt x 3kt frags:       12288 B  */
#define OFF_BUF    204800   /* f16 [16][10][128]:     40960 B  */
#define OFF_HL     245760   /* f32 [16][128]:          8192 B  */
#define OFF_WH32   253952   /* f32 [16][128]:          8192 B  */
#define OFF_WH16   262144   /* f16 [16][128]:          4096 B  */
#define OFF_INP16  266240   /* f16 [16][96]:           3072 B  */
#define OFF_OBS1   269312   /* f16 [16][160]:          5120 B  */
#define OFF_HID1A  274432   /* f32 [16][64]:           4096 B  */
#define OFF_HID1B  278528   /* f32 [16][64]:           4096 B  */
#define OFF_LOG    282624   /* f32 [2][16][10]:        1280 B  */
#define OFF_ACT    283904   /* int [2][16]:             128 B  */
#define OFF_BIH    284032   /* f32 [384]:              1536 B  */
#define OFF_BHH    285568   /* f32 [384]:              1536 B  */
#define OFF_B1A    287104   /* f32 [64]:                256 B  */
#define OFF_B1B    287360   /* f32 [64]:                256 B  */
#define OFF_DEMO   287616   /* f32 [16][2]:             128 B  */
#define SMEM_BYTES 287744

union U16h { v16h v; v8h h[2]; };

__device__ __forceinline__ unsigned hash_u32(unsigned x) {
    x ^= x >> 16; x *= 0x7feb352du;
    x ^= x >> 15; x *= 0x846ca68bu;
    x ^= x >> 16; return x;
}
__device__ __forceinline__ float sigmoidf(float x) { return 1.0f / (1.0f + __expf(-x)); }

// A-fragment loader: 16x32 f16 A matrix per ISA layout.
// lane 0-15 : row M=lane,    halfs 0-7 -> K kt*32+0..7,  halfs 8-15 -> K kt*32+16..23
// lane 16-31: row M=lane-16, halfs 0-7 -> K kt*32+8..15, halfs 8-15 -> K kt*32+24..31
__device__ __forceinline__ v16h loadA(const _Float16* X, int Kpad, int kt, int lane) {
    int m = lane & 15, hi = lane >> 4;
    const _Float16* base = X + m * Kpad + kt * 32 + hi * 8;
    U16h u;
    u.h[0] = *(const v8h*)(base);
    u.h[1] = *(const v8h*)(base + 16);
    return u.v;
}

// B-fragment loader: pre-swizzled fragments, lane-contiguous 32 B per lane.
__device__ __forceinline__ v16h loadB(const _Float16* F, int frag, int lane) {
    const _Float16* base = F + frag * 512 + lane * 16;
    U16h u;
    u.h[0] = *(const v8h*)(base);
    u.h[1] = *(const v8h*)(base + 8);
    return u.v;
}

__device__ __forceinline__ v8f wmma_f16(v16h a, v16h b, v8f c) {
    return __builtin_amdgcn_wmma_f32_16x16x32_f16(false, a, false, b, (short)0, c, false, false);
}

// Swizzle W ([N][Kin] row-major f32 in global) into B fragments (f16 in LDS).
// Fragment (nt,kt) index = nt*nFragsK + kt; 32x16 per fragment.
// lane<16:  N = nt*16+lane,    halfs 0-7 -> K kt*32+0..7,  halfs 8-15 -> K kt*32+16..23
// lane>=16: N = nt*16+lane-16, halfs 0-7 -> K kt*32+8..15, halfs 8-15 -> K kt*32+24..31
__device__ __forceinline__ void swizzleB(const float* W, int Kin, int nFragsK,
                                         int nFrags, _Float16* dst, int tid) {
    int total = nFrags * 512;
    for (int idx = tid; idx < total; idx += NTHREADS) {
        int frag = idx >> 9, el = idx & 511;
        int lane = el >> 4, half = el & 15;
        int nt = frag / nFragsK, kt = frag - nt * nFragsK;
        int n = nt * 16 + (lane & 15);
        int k = kt * 32 + ((half >> 3) << 4) + ((lane >> 4) << 3) + (half & 7);
        float v = (k < Kin) ? W[n * Kin + k] : 0.0f;
        dst[idx] = (_Float16)v;
    }
}

// One GRU step; wave w owns hidden columns [16w, 16w+16). r/z/n kept in regs.
// w_hh B-fragments (BR/BZ/BN, 12 x v16h) are register-resident across the scan.
__device__ __forceinline__ void gru_step(const v16h* BR, const v16h* BZ, const v16h* BN,
                                         const _Float16* WBIH,
                                         const _Float16* INP16, const _Float16* WH16,
                                         const float* WH32, float* HL,
                                         const float* BIH, const float* BHH,
                                         int lane, int w) {
    int cl = lane & 15, hi = lane >> 4;
    int col = (w << 4) + cl;
    float bir = BIH[col], biz = BIH[Hn + col], bin = BIH[2 * Hn + col];
    float bhr = BHH[col], bhz = BHH[Hn + col], bhn = BHH[2 * Hn + col];
    v8f ir, iz, inx, hr, hz, hn;
#pragma unroll
    for (int i = 0; i < 8; i++) { ir[i]=bir; iz[i]=biz; inx[i]=bin; hr[i]=bhr; hz[i]=bhz; hn[i]=bhn; }
#pragma unroll
    for (int kt = 0; kt < 3; kt++) {              // gi = inp @ w_ih^T (K padded 73->96)
        v16h a = loadA(INP16, 96, kt, lane);
        ir  = wmma_f16(a, loadB(WBIH, (w     ) * 3 + kt, lane), ir);
        iz  = wmma_f16(a, loadB(WBIH, (w +  8) * 3 + kt, lane), iz);
        inx = wmma_f16(a, loadB(WBIH, (w + 16) * 3 + kt, lane), inx);
    }
#pragma unroll
    for (int kt = 0; kt < 4; kt++) {              // gh = wh @ w_hh^T (K=128), weights in regs
        v16h a = loadA(WH16, 128, kt, lane);
        hr = wmma_f16(a, BR[kt], hr);
        hz = wmma_f16(a, BZ[kt], hz);
        hn = wmma_f16(a, BN[kt], hn);
    }
#pragma unroll
    for (int i = 0; i < 8; i++) {                 // gates in-register; D layout M=i+8*hi, N=cl
        int r = i + (hi << 3);
        float rg = sigmoidf(ir[i] + hr[i]);
        float zg = sigmoidf(iz[i] + hz[i]);
        float ng = tanhf(inx[i] + rg * hn[i]);
        float whv = WH32[r * Hn + col];
        HL[r * Hn + col] = (1.0f - zg) * ng + zg * whv;
    }
}

__global__ void __launch_bounds__(NTHREADS)
Agent_74878459838639_kernel(const float* __restrict__ x,
                            const float* __restrict__ w_init_h, const float* __restrict__ b_init_h,
                            const float* __restrict__ w_ih,     const float* __restrict__ b_ih,
                            const float* __restrict__ w_hh,     const float* __restrict__ b_hh,
                            const float* __restrict__ w1a,      const float* __restrict__ b1a,
                            const float* __restrict__ w2a,      const float* __restrict__ b2a,
                            const float* __restrict__ w1b,      const float* __restrict__ b1b,
                            const float* __restrict__ w2b,      const float* __restrict__ b2b,
                            const float* __restrict__ w_f,      const float* __restrict__ b_f,
                            float* __restrict__ out) {
    extern __shared__ __align__(16) char smem[];
    _Float16* WBHH  = (_Float16*)(smem + OFF_WBHH);
    _Float16* WBIH  = (_Float16*)(smem + OFF_WBIH);
    _Float16* W1A   = (_Float16*)(smem + OFF_W1A);
    _Float16* W1B   = (_Float16*)(smem + OFF_W1B);
    _Float16* BUF   = (_Float16*)(smem + OFF_BUF);
    float*    HL    = (float*)   (smem + OFF_HL);
    float*    WH32  = (float*)   (smem + OFF_WH32);
    _Float16* WH16  = (_Float16*)(smem + OFF_WH16);
    _Float16* INP16 = (_Float16*)(smem + OFF_INP16);
    _Float16* OBS1  = (_Float16*)(smem + OFF_OBS1);
    float*    HID1A = (float*)   (smem + OFF_HID1A);
    float*    HID1B = (float*)   (smem + OFF_HID1B);
    float*    LOGITS= (float*)   (smem + OFF_LOG);
    int*      ACT   = (int*)     (smem + OFF_ACT);
    float*    BIH   = (float*)   (smem + OFF_BIH);
    float*    BHH   = (float*)   (smem + OFF_BHH);
    float*    B1A   = (float*)   (smem + OFF_B1A);
    float*    B1B   = (float*)   (smem + OFF_B1B);
    float*    DEMOL = (float*)   (smem + OFF_DEMO);

    const int tid  = threadIdx.x;
    const int lane = tid & 31;
    const int w    = tid >> 5;          // wave id 0..7
    const int cl   = lane & 15;
    const int hi   = lane >> 4;
    const int row0 = blockIdx.x * 16;   // 512 blocks x 16 batch rows

    // ---------- one-time setup ----------
    for (int i = tid; i < 3 * Hn; i += NTHREADS) { BIH[i] = b_ih[i]; BHH[i] = b_hh[i]; }
    for (int i = tid; i < NUn; i += NTHREADS)    { B1A[i] = b1a[i];  B1B[i] = b1b[i]; }
    if (tid < 32) {
        int r = tid >> 1, c = tid & 1;
        DEMOL[r * 2 + c] = x[(size_t)(row0 + r) * Tt * (DEMOn + LABn) + c];
    }
    for (int i = tid; i < 16 * NAn * Hn; i += NTHREADS) BUF[i] = (_Float16)0.0f;

    swizzleB(w_hh, Hn,  4, 24 * 4, WBHH, tid);   // 96 frags
    swizzleB(w_ih, LABn, 3, 24 * 3, WBIH, tid);  // 72 frags, K zero-padded 73->96
    swizzleB(w1a, Hn + DEMOn, 5, 4 * 5, W1A, tid); // K=130 -> pad 160
    swizzleB(w1b, LABn + DEMOn, 3, 4 * 3, W1B, tid); // K=75 -> pad 96
    __syncthreads();

    // Hoist this wave's 12 loop-invariant w_hh B-fragments into registers:
    // removes ~43% of the per-step LDS traffic on the sequential critical path.
    v16h BR[4], BZ[4], BN[4];
#pragma unroll
    for (int kt = 0; kt < 4; kt++) {
        BR[kt] = loadB(WBHH, (w     ) * 4 + kt, lane);
        BZ[kt] = loadB(WBHH, (w +  8) * 4 + kt, lane);
        BN[kt] = loadB(WBHH, (w + 16) * 4 + kt, lane);
    }

    // demo-dependent staged constants (persist across steps)
    for (int i = tid; i < 16 * 160; i += NTHREADS) {     // OBS1 cols 128..159
        int r = i / 160, c = i - r * 160;
        if (c >= Hn) {
            float v = 0.0f;
            if (c == Hn)     v = DEMOL[r * 2 + 0];
            if (c == Hn + 1) v = DEMOL[r * 2 + 1];
            OBS1[i] = (_Float16)v;
        }
    }
    for (int i = tid; i < 16 * 96; i += NTHREADS) {      // INP16 cols 73..95
        int r = i / 96, c = i - r * 96;
        if (c >= LABn) {
            float v = 0.0f;
            if (c == LABn)     v = DEMOL[r * 2 + 0];
            if (c == LABn + 1) v = DEMOL[r * 2 + 1];
            INP16[i] = (_Float16)v;
        }
    }
    // h0 = demo @ w_init_h^T + b; staged as wh for the first GRU step
    for (int i = tid; i < 16 * Hn; i += NTHREADS) {
        int r = i >> 7, c = i & 127;
        float v = b_init_h[c] + DEMOL[r * 2] * w_init_h[c * 2] + DEMOL[r * 2 + 1] * w_init_h[c * 2 + 1];
        WH32[i] = v; WH16[i] = (_Float16)v;
    }
    // stage inp for t=0
    for (int i = tid; i < 16 * LABn; i += NTHREADS) {
        int r = i / LABn, c = i - r * LABn;
        INP16[r * 96 + c] = (_Float16)x[((size_t)(row0 + r) * Tt + 0) * (DEMOn + LABn) + DEMOn + c];
    }
    // prefetch t=1 rows into cache while the first GRU step runs
    if (tid < 16)
        __builtin_prefetch(&x[((size_t)(row0 + tid) * Tt + 1) * (DEMOn + LABn) + DEMOn], 0, 3);
    __syncthreads();

    gru_step(BR, BZ, BN, WBIH, INP16, WH16, WH32, HL, BIH, BHH, lane, w);
    __syncthreads();

    // ---------- scan over t = 1..47 ----------
    for (int t = 1; t < Tt; t++) {
        // push h into circular buffer slot (t-1)%10; stage inp[t]
        int slot = (t - 1) % NAn;
        for (int i = tid; i < 16 * Hn; i += NTHREADS) {
            int r = i >> 7, c = i & 127;
            BUF[(r * NAn + slot) * Hn + c] = (_Float16)HL[i];
        }
        for (int i = tid; i < 16 * LABn; i += NTHREADS) {
            int r = i / LABn, c = i - r * LABn;
            INP16[r * 96 + c] = (_Float16)x[((size_t)(row0 + r) * Tt + t) * (DEMOn + LABn) + DEMOn + c];
        }
        // prefetch next step's input rows (global_prefetch_b8)
        if (t + 1 < Tt && tid < 16)
            __builtin_prefetch(&x[((size_t)(row0 + tid) * Tt + t + 1) * (DEMOn + LABn) + DEMOn], 0, 3);
        __syncthreads();

        // obs1 = [mean(buf), demo]
        for (int i = tid; i < 16 * Hn; i += NTHREADS) {
            int r = i >> 7, c = i & 127;
            float s = 0.0f;
#pragma unroll
            for (int q = 0; q < NAn; q++) s += (float)BUF[(r * NAn + q) * Hn + c];
            OBS1[r * 160 + c] = (_Float16)(s * 0.1f);
        }
        __syncthreads();

        // sampler layer-1 via WMMA: waves 0-3 -> A (K pad 160), waves 4-7 -> B (K pad 96)
        {
            v8f acc;
            if (w < 4) {
                float bb = B1A[(w << 4) + cl];
#pragma unroll
                for (int i = 0; i < 8; i++) acc[i] = bb;
#pragma unroll
                for (int kt = 0; kt < 5; kt++)
                    acc = wmma_f16(loadA(OBS1, 160, kt, lane), loadB(W1A, w * 5 + kt, lane), acc);
#pragma unroll
                for (int i = 0; i < 8; i++)
                    HID1A[(i + (hi << 3)) * NUn + (w << 4) + cl] = tanhf(acc[i]);
            } else {
                int nt = w - 4;
                float bb = B1B[(nt << 4) + cl];
#pragma unroll
                for (int i = 0; i < 8; i++) acc[i] = bb;
#pragma unroll
                for (int kt = 0; kt < 3; kt++)
                    acc = wmma_f16(loadA(INP16, 96, kt, lane), loadB(W1B, nt * 3 + kt, lane), acc);
#pragma unroll
                for (int i = 0; i < 8; i++)
                    HID1B[(i + (hi << 3)) * NUn + (nt << 4) + cl] = tanhf(acc[i]);
            }
        }
        __syncthreads();

        // layer-2 (64 -> 10), scalar: 320 dot products
        for (int task = tid; task < 2 * 16 * NAn; task += NTHREADS) {
            int s = task / 160, rem = task - s * 160;
            int rr = rem / NAn, c = rem - rr * NAn;
            const float* hsrc = s ? HID1B : HID1A;
            const float* W2   = s ? w2b : w2a;
            const float* Bi2  = s ? b2b : b2a;
            float acc = Bi2[c];
#pragma unroll 8
            for (int k = 0; k < NUn; k++) acc += hsrc[rr * NUn + k] * W2[c * NUn + k];
            LOGITS[(s * 16 + rr) * NAn + c] = acc;
        }
        __syncthreads();

        // Gumbel-argmax categorical sample (deterministic stateless hash RNG)
        if (tid < 32) {
            int s = tid >> 4, rr = tid & 15;
            unsigned base = (unsigned)(row0 + rr) * 2654435761u ^ ((unsigned)t * 40503u + (unsigned)s * 2246822519u);
            float best = -1e30f; int bi = 0;
            for (int c = 0; c < NAn; c++) {
                unsigned hsh = hash_u32(base + (unsigned)c * 374761393u);
                float u = (float)(hsh & 0xFFFFFFu) * (1.0f / 16777216.0f) + 1e-7f;
                float g = -__logf(-__logf(u));
                float v = LOGITS[(s * 16 + rr) * NAn + c] + g;
                if (v > best) { best = v; bi = c; }
            }
            ACT[s * 16 + rr] = bi;
        }
        __syncthreads();

        // wh = LAMDA*0.5*(buf[a1]+buf[a2]) + (1-LAMDA)*h ; logical a -> physical (t+a)%10
        for (int i = tid; i < 16 * Hn; i += NTHREADS) {
            int r = i >> 7, c = i & 127;
            int s1 = (t + ACT[r]) % NAn, s2 = (t + ACT[16 + r]) % NAn;
            float ah = 0.5f * ((float)BUF[(r * NAn + s1) * Hn + c] + (float)BUF[(r * NAn + s2) * Hn + c]);
            float whv = 0.5f * ah + 0.5f * HL[i];
            WH32[i] = whv; WH16[i] = (_Float16)whv;
        }
        __syncthreads();

        gru_step(BR, BZ, BN, WBIH, INP16, WH16, WH32, HL, BIH, BHH, lane, w);
        __syncthreads();
    }

    // ---------- final projection: relu([h, demo] @ w_f^T + b_f) ----------
    swizzleB(w_f, Hn + DEMOn, 5, 8 * 5, WBIH, tid);   // reuse w_ih fragment space
    for (int i = tid; i < 16 * Hn; i += NTHREADS) {   // OBS1 cols 0..127 = h (128/129 = demo persist)
        int r = i >> 7, c = i & 127;
        OBS1[r * 160 + c] = (_Float16)HL[i];
    }
    __syncthreads();

    {
        int col = (w << 4) + cl;
        float bb = b_f[col];
        v8f acc;
#pragma unroll
        for (int i = 0; i < 8; i++) acc[i] = bb;
#pragma unroll
        for (int kt = 0; kt < 5; kt++)
            acc = wmma_f16(loadA(OBS1, 160, kt, lane), loadB(WBIH, w * 5 + kt, lane), acc);
#pragma unroll
        for (int i = 0; i < 8; i++) {
            int r = row0 + i + (hi << 3);
            out[(size_t)r * FDn + col] = fmaxf(acc[i], 0.0f);
        }
    }
}

extern "C" void kernel_launch(void* const* d_in, const int* in_sizes, int n_in,
                              void* d_out, int out_size, void* d_ws, size_t ws_size,
                              hipStream_t stream) {
    (void)in_sizes; (void)n_in; (void)d_ws; (void)ws_size; (void)out_size;
    const float* x        = (const float*)d_in[0];
    const float* w_init_h = (const float*)d_in[1];
    const float* b_init_h = (const float*)d_in[2];
    const float* w_ih     = (const float*)d_in[3];
    const float* b_ih     = (const float*)d_in[4];
    const float* w_hh     = (const float*)d_in[5];
    const float* b_hh     = (const float*)d_in[6];
    const float* w1a      = (const float*)d_in[7];
    const float* b1a      = (const float*)d_in[8];
    const float* w2a      = (const float*)d_in[9];
    const float* b2a      = (const float*)d_in[10];
    const float* w1b      = (const float*)d_in[11];
    const float* b1b      = (const float*)d_in[12];
    const float* w2b      = (const float*)d_in[13];
    const float* b2b      = (const float*)d_in[14];
    const float* w_f      = (const float*)d_in[15];
    const float* b_f      = (const float*)d_in[16];
    float* out = (float*)d_out;

    (void)hipFuncSetAttribute((const void*)Agent_74878459838639_kernel,
                              hipFuncAttributeMaxDynamicSharedMemorySize, SMEM_BYTES);
    dim3 grid(Bb / 16), block(NTHREADS);
    Agent_74878459838639_kernel<<<grid, block, SMEM_BYTES, stream>>>(
        x, w_init_h, b_init_h, w_ih, b_ih, w_hh, b_hh,
        w1a, b1a, w2a, b2a, w1b, b1b, w2b, b2b, w_f, b_f, out);
}
